// IrrepWiseLinear_7232724927037
// MI455X (gfx1250) — compile-verified, matrix operand order checked by reading, not run on hardware
//
#include <hip/hip_runtime.h>
#include <hip/hip_bf16.h>

typedef __bf16 v16bf __attribute__((ext_vector_type(16)));
typedef __bf16 v8bf  __attribute__((ext_vector_type(8)));
typedef float  v8f   __attribute__((ext_vector_type(8)));
typedef float  v4f   __attribute__((ext_vector_type(4)));

#define DIM   16
#define CIN   128
#define COUT  128

// ---------------------------------------------------------------------------
// Prep: weight[p][k][n] (f32, 4x128x128) -> wt[p][n][k] (bf16, transposed)
// so that B-operand fragments become contiguous 16-byte loads.
// ---------------------------------------------------------------------------
__global__ __launch_bounds__(256) void wprep_kernel(const float* __restrict__ w,
                                                    __bf16* __restrict__ wt) {
    int o = blockIdx.x * blockDim.x + threadIdx.x;   // 0 .. 65535
    int k = o & 127;
    int n = (o >> 7) & 127;
    int p = o >> 14;
    wt[o] = (__bf16)w[(p << 14) + (k << 7) + n];
}

// ---------------------------------------------------------------------------
// Main: one workgroup = 16 samples x one spherical row m  (single path id)
//   A tile: x[nblk*16 .. +15][m][0..127]  (16x128)  -> bf16 LDS
//   8 waves, wave w computes output columns [w*16, w*16+16)
//   K-loop: 4 x v_wmma_f32_16x16x32_bf16
// ---------------------------------------------------------------------------
__global__ __launch_bounds__(256) void irrep_linear_kernel(
        const float* __restrict__ x,
        const __bf16* __restrict__ wt,
        float* __restrict__ out) {

    constexpr int LDA = 136;                 // bf16 units; 272B row stride -> bank spread
    __shared__ __bf16 a_lds[16 * LDA];

    const int m    = blockIdx.x & 15;        // spherical row
    const int nblk = blockIdx.x >> 4;        // block of 16 samples
    const int p    = (m >= 1) + (m >= 4) + (m >= 9);   // path id for this row

    const int t = threadIdx.x;               // 0..255

    // ---- stage A tile (fp32 -> bf16) into LDS ----
    {
        const int i  = t >> 4;               // sample-in-block 0..15
        const int k0 = (t & 15) << 3;        // 0,8,...,120
        const float* src = x + (((size_t)(nblk * 16 + i)) * DIM + m) * CIN + k0;
        v4f f0 = *(const v4f*)(src);
        v4f f1 = *(const v4f*)(src + 4);
        v8bf b;
        b[0] = (__bf16)f0[0]; b[1] = (__bf16)f0[1];
        b[2] = (__bf16)f0[2]; b[3] = (__bf16)f0[3];
        b[4] = (__bf16)f1[0]; b[5] = (__bf16)f1[1];
        b[6] = (__bf16)f1[2]; b[7] = (__bf16)f1[3];
        *(v8bf*)(&a_lds[i * LDA + k0]) = b;  // ds_store_b128
    }
    __syncthreads();

    const int wave    = t >> 5;              // 0..7 -> column tile
    const int lane    = t & 31;
    const int halfsel = lane >> 4;           // 0 | 1
    const int lrow    = lane & 15;

    // A fragment base (bf16 units): row = lane&15, K chunk starts at halfsel*8
    const int a_off = lrow * LDA + (halfsel << 3);
    // B fragment base: column n = wave*16 + lrow of wt[p][n][k]
    const __bf16* bbase = wt + (((size_t)p << 14) + ((size_t)(wave * 16 + lrow) << 7))
                             + (halfsel << 3);

    union V16 { v16bf v; v8bf h[2]; };

    v8f acc = {0.f, 0.f, 0.f, 0.f, 0.f, 0.f, 0.f, 0.f};

#pragma unroll
    for (int kb = 0; kb < 4; ++kb) {
        const int ka = kb * 32;
        V16 a, b;
        a.h[0] = *(const v8bf*)(&a_lds[a_off + ka]);        // K = base .. base+7
        a.h[1] = *(const v8bf*)(&a_lds[a_off + ka + 16]);   // K = base+16 .. base+23
        b.h[0] = *(const v8bf*)(bbase + ka);
        b.h[1] = *(const v8bf*)(bbase + ka + 16);
        acc = __builtin_amdgcn_wmma_f32_16x16x32_bf16(
                  /*neg_a=*/false, a.v, /*neg_b=*/false, b.v,
                  /*c_mod=*/(short)0, acc, /*reuse_a=*/false, /*reuse_b=*/false);
    }

    // ---- store D: VGPR r holds (M = r + 8*halfsel, N = lrow) of the 16x16 tile
    const int col = wave * 16 + lrow;
    float* obase = out + (((size_t)(nblk * 16)) * DIM + m) * COUT + col;
    const int mo = halfsel * 8;
#pragma unroll
    for (int r = 0; r < 8; ++r) {
        obase[(size_t)(mo + r) * (DIM * COUT)] = acc[r];
    }
}

// ---------------------------------------------------------------------------
extern "C" void kernel_launch(void* const* d_in, const int* in_sizes, int n_in,
                              void* d_out, int out_size, void* d_ws, size_t ws_size,
                              hipStream_t stream) {
    const float* x = (const float*)d_in[0];          // [N,16,128] f32
    const float* w = (const float*)d_in[1];          // [4,128,128] f32
    float*       out = (float*)d_out;                // [N,16,128] f32
    __bf16*      wt  = (__bf16*)d_ws;                // [4,128,128] bf16 (transposed)

    // convert + transpose weights (65536 elements)
    wprep_kernel<<<256, 256, 0, stream>>>(w, wt);

    const int n = in_sizes[0] / (DIM * CIN);         // number of samples
    const int grid = (n / 16) * 16;                  // (n/16) sample-blocks x 16 rows
    irrep_linear_kernel<<<grid, 256, 0, stream>>>(x, wt, out);
}
